// NaiveCustomLSTM_74491912782280
// MI455X (gfx1250) — compile-verified, matrix operand order checked by reading, not run on hardware
//
#include <hip/hip_runtime.h>
#include <hip/hip_bf16.h>
#include <math.h>

// ---------------------------------------------------------------------------
// LSTM: B=64, S=2048, H=512. out = [S,B,H] ++ h_T[B,H] ++ c_T[B,H]  (fp32)
//
// MI455X (gfx1250, wave32) strategy:
//  * 4 persistent workgroups (16-row batch groups; M=16 == WMMA tile height).
//  * z = [x_t | h] (16 x 1024 bf16) in LDS; weights [U;V] for all 4 gates
//    packed fragment-major bf16 in d_ws (4 MB -> L2 resident).
//  * 32 waves/WG; each wave owns 16 hidden cols x 4 gates, so the cell
//    update c = f*c + i*g is wave-local; c lives in a v8f register fragment.
//  * v_wmma_f32_16x16x32_bf16; 4 gate WMMAs share one A fragment per k-step.
//  * Streaming traffic (x in, out stores) uses non-temporal hints so the
//    4 MB W working set is not evicted from the 192 MB L2 between steps.
// ---------------------------------------------------------------------------

typedef __attribute__((ext_vector_type(16))) __bf16 v16bf;
typedef __attribute__((ext_vector_type(8)))  __bf16 v8bf;
typedef __attribute__((ext_vector_type(8)))  float  v8f;
typedef __attribute__((ext_vector_type(4)))  float  v4f;

#define LSTM_B 64
#define LSTM_S 2048
#define LSTM_H 512
#define LSTM_K 1024          // 2*H : concat [x | h]
#define KTILES 32            // K / 32
// packed W layout: frag(ntile 0..127, ktile 0..31, lane 0..31) = 16 bf16
//   offset = ntile*16384 + ktile*512 + lane*16   (elements)
#define NT_STRIDE   16384
#define KT_STRIDE   512
#define GATE_STRIDE (32 * NT_STRIDE)   // 32 ntiles per gate (512 cols)

// ---------------------------------------------------------------------------
// Pack [U_g ; V_g] (fp32 row-major, [H][H]) into fragment-major bf16 W.
// One thread per lane-fragment (16 bf16 = 32 B).
// ---------------------------------------------------------------------------
__global__ void lstm_pack_w(const float* __restrict__ Uf, const float* __restrict__ Vf,
                            const float* __restrict__ Ui, const float* __restrict__ Vi,
                            const float* __restrict__ Uo, const float* __restrict__ Vo,
                            const float* __restrict__ Uc, const float* __restrict__ Vc,
                            __bf16* __restrict__ W) {
    int idx   = blockIdx.x * blockDim.x + threadIdx.x;   // 0 .. 131071
    int lane  = idx & 31;
    int ktile = (idx >> 5) & 31;
    int ntile = idx >> 10;                               // 0..127
    int n     = ntile * 16 + (lane & 15);                // global gate column
    int gate  = n >> 9;                                  // 0:f 1:i 2:o 3:c
    int j     = n & 511;
    const float* U = (gate == 0) ? Uf : (gate == 1) ? Ui : (gate == 2) ? Uo : Uc;
    const float* V = (gate == 0) ? Vf : (gate == 1) ? Vi : (gate == 2) ? Vo : Vc;
    int lo = (lane < 16) ? 0 : 8;

    v16bf frag;
#pragma unroll
    for (int e = 0; e < 16; ++e) {
        int kk = (e < 8) ? (lo + e) : (16 + lo + (e - 8));   // B-frag K pattern
        int k  = ktile * 32 + kk;
        float v = (k < LSTM_H) ? U[k * LSTM_H + j] : V[(k - LSTM_H) * LSTM_H + j];
        frag[e] = (__bf16)v;
    }
    *(v16bf*)(W + (size_t)idx * 16) = frag;
}

// ---------------------------------------------------------------------------
// Persistent LSTM kernel: grid = 4 (batch groups of 16), block = 1024 (32 waves)
// ---------------------------------------------------------------------------
__launch_bounds__(1024, 1)
__global__ void lstm_scan(const float* __restrict__ x,
                          const float* __restrict__ bf, const float* __restrict__ bi,
                          const float* __restrict__ bo, const float* __restrict__ bc,
                          const __bf16* __restrict__ W,
                          float* __restrict__ out) {
    __shared__ __bf16 z[16 * LSTM_K];          // [row 0..15][col 0..1023], 32 KB

    const int tid   = threadIdx.x;
    const int lane  = tid & 31;
    const int wv    = tid >> 5;                // wave 0..31 -> hidden cols j0..j0+15
    const int b0    = blockIdx.x * 16;         // batch group base row
    const int j0    = wv * 16;
    const int nlo   = lane & 15;               // N within tile / A-row M
    const int mbase = (lane < 16) ? 0 : 8;     // C-frag row base & A-frag K-lo

    // per-lane biases (column = j0 + nlo of each gate)
    const float bias0 = bf[j0 + nlo];
    const float bias1 = bi[j0 + nlo];
    const float bias2 = bo[j0 + nlo];
    const float bias3 = bc[j0 + nlo];

    // h(0) = 0  -> zero the h half of z
    for (int i = tid; i < 16 * LSTM_H; i += 1024) {
        int r = i >> 9, c = i & 511;
        z[r * LSTM_K + LSTM_H + c] = (__bf16)0.0f;
    }

    v8f c_st  = {};                            // cell state fragment
    v8f h_lst = {};

    // x-loader addressing: 8 floats / thread -> 16 x 512 slice
    const int xr = tid >> 6;                   // row 0..15
    const int xc = (tid & 63) * 8;             // col 0..504
    const float* xsrc = x + ((size_t)(b0 + xr) * LSTM_S) * LSTM_H + xc;

    // per-wave W bases (gate g tile: ntile = g*32 + wv)
    const __bf16* w0 = W + (size_t)wv * NT_STRIDE + (size_t)lane * 16;
    const __bf16* w1 = w0 + 1 * GATE_STRIDE;
    const __bf16* w2 = w0 + 2 * GATE_STRIDE;
    const __bf16* w3 = w0 + 3 * GATE_STRIDE;

    for (int t = 0; t < LSTM_S; ++t) {
        // ---- stage x_t (bf16) into z[:, 0:512]; NT: read-once stream -----
        {
            const float* s = xsrc + (size_t)t * LSTM_H;
            v4f a0 = __builtin_nontemporal_load((const v4f*)(s));
            v4f a1 = __builtin_nontemporal_load((const v4f*)(s + 4));
            v8bf pk;
            pk[0] = (__bf16)a0.x; pk[1] = (__bf16)a0.y;
            pk[2] = (__bf16)a0.z; pk[3] = (__bf16)a0.w;
            pk[4] = (__bf16)a1.x; pk[5] = (__bf16)a1.y;
            pk[6] = (__bf16)a1.z; pk[7] = (__bf16)a1.w;
            *(v8bf*)&z[xr * LSTM_K + xc] = pk;
            if (t + 1 < LSTM_S)                // hide next step's x stream
                __builtin_prefetch(s + LSTM_H, 0, 0);
        }
        __syncthreads();                       // z = [x_t | h_{t-1}] complete

        // ---- fused GEMM: [16 x 1024] @ [1024 x (4 gate tiles)] -----------
        v8f a0 = {}, a1 = {}, a2 = {}, a3 = {};
#pragma unroll 4
        for (int kt = 0; kt < KTILES; ++kt) {
            union { v16bf v; v8bf h[2]; } af;  // shared A fragment
            const __bf16* zr = &z[nlo * LSTM_K + kt * 32 + mbase];
            af.h[0] = *(const v8bf*)(zr);
            af.h[1] = *(const v8bf*)(zr + 16);

            const size_t fo = (size_t)kt * KT_STRIDE;
            v16bf bw0 = *(const v16bf*)(w0 + fo);   // RT policy: keep in L2
            v16bf bw1 = *(const v16bf*)(w1 + fo);
            v16bf bw2 = *(const v16bf*)(w2 + fo);
            v16bf bw3 = *(const v16bf*)(w3 + fo);

            a0 = __builtin_amdgcn_wmma_f32_16x16x32_bf16(false, af.v, false, bw0, (short)0, a0, false, false);
            a1 = __builtin_amdgcn_wmma_f32_16x16x32_bf16(false, af.v, false, bw1, (short)0, a1, false, false);
            a2 = __builtin_amdgcn_wmma_f32_16x16x32_bf16(false, af.v, false, bw2, (short)0, a2, false, false);
            a3 = __builtin_amdgcn_wmma_f32_16x16x32_bf16(false, af.v, false, bw3, (short)0, a3, false, false);
        }

        // ---- gate activations + cell update (wave-local) -----------------
        v8f hv;
#pragma unroll
        for (int e = 0; e < 8; ++e) {
            float fg = 1.0f / (1.0f + __expf(-(a0[e] + bias0)));
            float ig = 1.0f / (1.0f + __expf(-(a1[e] + bias1)));
            float og = 1.0f / (1.0f + __expf(-(a2[e] + bias2)));
            float gg = tanhf(a3[e] + bias3);
            float cc = fg * c_st[e] + ig * gg;
            c_st[e]  = cc;
            hv[e]    = og * tanhf(cc);
        }
        h_lst = hv;

        __syncthreads();                       // all A-reads of z done

        // ---- publish h_t: LDS (bf16, next step) + global out (NT fp32) ---
#pragma unroll
        for (int e = 0; e < 8; ++e) {
            int m = mbase + e;                 // C-frag row -> batch row
            z[m * LSTM_K + LSTM_H + j0 + nlo] = (__bf16)hv[e];
            __builtin_nontemporal_store(
                hv[e],
                &out[(size_t)t * (LSTM_B * LSTM_H) + (size_t)(b0 + m) * LSTM_H + j0 + nlo]);
        }
    }

    // ---- final (h_T, c_T) ------------------------------------------------
    float* hT = out + (size_t)LSTM_S * LSTM_B * LSTM_H;
    float* cT = hT + LSTM_B * LSTM_H;
#pragma unroll
    for (int e = 0; e < 8; ++e) {
        int m = mbase + e;
        size_t o = (size_t)(b0 + m) * LSTM_H + j0 + nlo;
        hT[o] = h_lst[e];
        cT[o] = c_st[e];
    }
}

// ---------------------------------------------------------------------------
extern "C" void kernel_launch(void* const* d_in, const int* in_sizes, int n_in,
                              void* d_out, int out_size, void* d_ws, size_t ws_size,
                              hipStream_t stream) {
    (void)in_sizes; (void)n_in; (void)out_size; (void)ws_size;
    const float* x  = (const float*)d_in[0];
    const float* Uf = (const float*)d_in[1];
    const float* Vf = (const float*)d_in[2];
    const float* bfv= (const float*)d_in[3];
    const float* Ui = (const float*)d_in[4];
    const float* Vi = (const float*)d_in[5];
    const float* biv= (const float*)d_in[6];
    const float* Uo = (const float*)d_in[7];
    const float* Vo = (const float*)d_in[8];
    const float* bov= (const float*)d_in[9];
    const float* Uc = (const float*)d_in[10];
    const float* Vc = (const float*)d_in[11];
    const float* bcv= (const float*)d_in[12];
    float*  out = (float*)d_out;
    __bf16* W   = (__bf16*)d_ws;               // 1024*2048 bf16 = 4 MB

    // pack weights: 128 ntiles * 32 ktiles * 32 lanes = 131072 fragments
    lstm_pack_w<<<512, 256, 0, stream>>>(Uf, Vf, Ui, Vi, Uo, Vo, Uc, Vc, W);

    // persistent scan: 4 batch groups x 32 waves
    lstm_scan<<<LSTM_B / 16, 1024, 0, stream>>>(x, bfv, biv, bov, bcv, W, out);
}
